// Corr2D_2662879723528
// MI455X (gfx1250) — matrix operand order repaired and checked by reading, not compiled, not run on hardware
//
#include <hip/hip_runtime.h>

#define B_  8
#define C_  256
#define H_  128
#define W_  128
#define MD  4
#define ND  9            // 2*MD+1 displacements per axis
#define TILE_W 32        // output pixels per workgroup along W
#define NWT  (W_ / TILE_W)
#define JMAX 48          // in2 window: TILE_W + 16
#define CP   264         // padded channel pitch (bf16 elems): 528B rows, 16B aligned, bank-spread
#define NTHREADS 128     // 4 waves
#define SLAB_C 64        // channels per wave for TDM staging

// dynamic LDS layout (bytes)
#define STAGE_BYTES (4 * SLAB_C * JMAX * 4)          // 4 waves * [64][48] f32 = 49152
#define LDS1_OFF    STAGE_BYTES                      // in1 bf16 [32][CP]   = 16896
#define LDS2_OFF    (LDS1_OFF + TILE_W * CP * 2)     // in2 bf16 [48][CP]   = 25344
#define SMEM_BYTES  (LDS2_OFF + JMAX * CP * 2)       // 91392 total

typedef __attribute__((ext_vector_type(16))) __bf16 v16bf;
typedef __attribute__((ext_vector_type(8)))  __bf16 v8bf;
typedef __attribute__((ext_vector_type(8)))  float  v8f;
typedef __attribute__((ext_vector_type(4)))  unsigned int u32x4;
typedef __attribute__((ext_vector_type(8)))  int  i32x8;
typedef __attribute__((ext_vector_type(4)))  int  i32x4;

__device__ __forceinline__ unsigned short f2bf(float f) {
  unsigned u = __builtin_bit_cast(unsigned, f);
  u += 0x7FFFu + ((u >> 16) & 1u);          // round-to-nearest-even
  return (unsigned short)(u >> 16);
}

// TDM: DMA a 2D tile (tile_dim0=48 pixels x tile_dim1=64 channels, f32) from
// global to LDS. D# packing per CDNA5 ISA ch.8 (group0 128b, group1 256b).
__device__ __forceinline__ void tdm_load_tile(unsigned lds_off, const float* g,
                                              int valid_w) {
  unsigned long long ga = (unsigned long long)(size_t)g;
  u32x4 g0;
  g0[0] = 1u;                                           // count=1, user desc
  g0[1] = lds_off;                                      // lds_addr [63:32]
  g0[2] = (unsigned)ga;                                 // global_addr lo
  g0[3] = (unsigned)((ga >> 32) & 0x01FFFFFFu)          // global_addr [56:32]
        | (2u << 30);                                   // type=2 ("image")
  i32x8 g1;
  g1[0] = 0x00020000;                                   // data_size=2 (4B)
  g1[1] = (valid_w & 0xFFFF) << 16;                     // tensor_dim0 lo16 (right-edge OOB->0)
  g1[2] = (SLAB_C << 16);                               // tensor_dim0 hi=0 | tensor_dim1=64
  g1[3] = (JMAX << 16);                                 // tensor_dim1 hi=0 | tile_dim0=48
  g1[4] = SLAB_C;                                       // tile_dim1=64, tile_dim2=0
  g1[5] = H_ * W_;                                      // tensor_dim0_stride = 16384 (channel stride)
  g1[6] = 0;
  g1[7] = 0;
  i32x4 g2 = {0, 0, 0, 0};                              // 2D: dims 2/3 unused
  i32x4 g3 = {0, 0, 0, 0};
  i32x8 g4 = {0, 0, 0, 0, 0, 0, 0, 0};                  // unused VADDR4 slot (6-arg form)
  __builtin_amdgcn_tensor_load_to_lds(g0, g1, g2, g3, g4, 0);
}

__global__ __launch_bounds__(NTHREADS)
void corr81_wmma(const float* __restrict__ in1,
                 const float* __restrict__ in2,
                 float* __restrict__ out) {
  extern __shared__ char smem[];
  float*  stage = (float*)smem;                          // [wave][SLAB_C][JMAX] f32
  __bf16* lds1  = (__bf16*)(smem + LDS1_OFF);            // in1 [pixel][channel] bf16
  __bf16* lds2  = (__bf16*)(smem + LDS2_OFF);            // in2 [pixel][channel] bf16
  unsigned short* s1 = (unsigned short*)lds1;
  unsigned short* s2 = (unsigned short*)lds2;

  const int tid = threadIdx.x;
  const int bid = blockIdx.x;
  const int wt  = bid & (NWT - 1);
  const int h   = (bid / NWT) & (H_ - 1);
  const int b   = bid / (NWT * H_);
  const int w0  = wt * TILE_W;

  // ---- load in1 tile (fp32 -> bf16), pixel-major; fully coalesced ----
  for (int i = tid; i < TILE_W * C_; i += NTHREADS) {
    int c  = i >> 5;
    int wl = i & (TILE_W - 1);
    float v = in1[(((b * C_ + c) * H_ + h) << 7) + w0 + wl];
    s1[wl * CP + c] = f2bf(v);
  }

  const int lane = tid & 31;
  const int wv   = tid >> 5;         // wave 0..3
  const int sS   = wv >> 1;          // in1 subtile (0/1)
  const int tT   = wv & 1;           // A-tile within 24-pixel window
  const int n    = lane & 15;
  const int hh   = lane >> 4;

  const __bf16* brow = &lds1[(16 * sS + n) * CP + 8 * hh];
  const __bf16* arow = &lds2[(16 * (sS + tT) + n) * CP + 8 * hh];
  const int wout = w0 + 16 * sS + n;

  // ---- TDM staging geometry (clamp left edge; tensor_dim0 zeroes right edge) ----
  const int gw     = (wt == 0) ? 0 : (w0 - MD);   // clamped global start pixel
  const int shift  = gw - (w0 - MD);              // 0, or 4 when wt==0
  const int validw = W_ - gw;
  const float* slabbase = in2 + ((size_t)(b * C_ + SLAB_C * wv) * H_) * W_ + gw;
  const unsigned stage_off = (unsigned)(size_t)(stage + wv * (SLAB_C * JMAX));
  const float* myslab = stage + wv * (SLAB_C * JMAX);

  // prime the pipeline: TDM for dy=0 flies while we finish the in1 load
  {
    int hy0 = h - MD;
    if ((unsigned)hy0 < (unsigned)H_)
      tdm_load_tile(stage_off, slabbase + (size_t)hy0 * W_, validw);
  }

  for (int dy = 0; dy < ND; ++dy) {
    __syncthreads();                 // previous compute done -> lds2 writable

    // ---- convert this wave's staged slab f32 -> bf16, [c][x] -> [jl][c] ----
    const int hy = h + dy - MD;
    const bool hyok = (unsigned)hy < (unsigned)H_;
    __builtin_amdgcn_s_wait_tensorcnt(0);
#pragma unroll 4
    for (int cl = 0; cl < SLAB_C; ++cl) {
      const int cg = SLAB_C * wv + cl;
      {
        int jl = lane;               // jl 0..31
        int x  = jl - shift;
        float v = (hyok && (unsigned)x < (unsigned)JMAX) ? myslab[cl * JMAX + x] : 0.0f;
        s2[jl * CP + cg] = f2bf(v);
      }
      if (lane < 16) {
        int jl = lane + 32;          // jl 32..47
        int x  = jl - shift;
        float v = (hyok && (unsigned)x < (unsigned)JMAX) ? myslab[cl * JMAX + x] : 0.0f;
        s2[jl * CP + cg] = f2bf(v);
      }
    }

    // ---- staging consumed: launch TDM for dy+1 so it overlaps the WMMAs ----
    if (dy + 1 < ND) {
      int hyn = h + dy + 1 - MD;
      if ((unsigned)hyn < (unsigned)H_) {
        asm volatile("s_wait_dscnt 0" ::: "memory");   // convert's LDS reads done
        tdm_load_tile(stage_off, slabbase + (size_t)hyn * W_, validw);
      }
    }
    __syncthreads();

    // ---- one 16x16 GEMM chain over K = 256 (8x v_wmma_f32_16x16x32_bf16) ----
    v8f acc = {};
#pragma unroll
    for (int kk = 0; kk < C_ / 32; ++kk) {
      const int k0 = kk * 32;
      v8bf b_lo = *(const v8bf*)(brow + k0);
      v8bf b_hi = *(const v8bf*)(brow + k0 + 16);
      v16bf bf = __builtin_shufflevector(b_lo, b_hi,
                   0,1,2,3,4,5,6,7,8,9,10,11,12,13,14,15);
      v8bf a_lo = *(const v8bf*)(arow + k0);
      v8bf a_hi = *(const v8bf*)(arow + k0 + 16);
      v16bf af = __builtin_shufflevector(a_lo, a_hi,
                   0,1,2,3,4,5,6,7,8,9,10,11,12,13,14,15);
      acc = __builtin_amdgcn_wmma_f32_16x16x32_bf16(
                false, af, false, bf, (short)0, acc, false, false);
    }

    // ---- extract the |dx|<=4 band diagonals and store ----
#pragma unroll
    for (int v = 0; v < 8; ++v) {
      int M    = v + 8 * hh;
      int dxp4 = 16 * tT + M - n;          // dx + 4
      if (dxp4 >= 0 && dxp4 < ND) {
        int d = dy * ND + dxp4;
        out[(((b * 81 + d) * H_ + h) << 7) + wout] = acc[v] * (1.0f / 256.0f);
      }
    }
  }
}

extern "C" void kernel_launch(void* const* d_in, const int* in_sizes, int n_in,
                              void* d_out, int out_size, void* d_ws, size_t ws_size,
                              hipStream_t stream) {
  (void)in_sizes; (void)n_in; (void)out_size; (void)d_ws; (void)ws_size;
  const float* in1 = (const float*)d_in[0];
  const float* in2 = (const float*)d_in[1];
  float* out = (float*)d_out;
  dim3 grid(B_ * H_ * NWT);                // 8 * 128 * 4 = 4096 workgroups
  corr81_wmma<<<grid, NTHREADS, SMEM_BYTES, stream>>>(in1, in2, out);
}
